// MultiscaleDeformableSelfAttention_48979807043713
// MI455X (gfx1250) — matrix operand order repaired and checked by reading, not compiled, not run on hardware
//
#include <hip/hip_runtime.h>
#include <hip/hip_bf16.h>

// ---------------- problem constants (fixed by the reference) ----------------
#define N_     4
#define NQ_    5440
#define EMBED_ 256
#define HEADS_ 8
#define LEVELS_ 4
#define POINTS_ 4
#define DH_    32
#define M_     (N_ * NQ_)          // 21760 rows; M/16 = 1360 tiles, divisible by 8

typedef __bf16 bf16;
typedef __attribute__((ext_vector_type(8)))  bf16  v8bf;
typedef __attribute__((ext_vector_type(16))) bf16  v16bf;
typedef __attribute__((ext_vector_type(8)))  float v8f;

#define BPAD 264   // LDS row stride in elements (528B): bank-conflict-free b128 reads

// Build a 16-wide bf16 fragment from two 8-wide register chunks.
__device__ __forceinline__ v16bf cat16(v8bf lo, v8bf hi) {
    return __builtin_shufflevector(lo, hi, 0,1,2,3,4,5,6,7,8,9,10,11,12,13,14,15);
}

// gfx1250 async copy: global -> LDS, 16B per lane, tracked by ASYNCcnt.
__device__ __forceinline__ void async_ld_b128(unsigned lds_byte_addr, const void* gptr) {
    asm volatile("global_load_async_to_lds_b128 %0, %1, off"
                 :: "v"(lds_byte_addr), "v"(gptr) : "memory");
}
__device__ __forceinline__ void wait_asynccnt0() {
    asm volatile("s_wait_asynccnt 0x0" ::: "memory");
}

// ---------------- prep kernels ----------------
// hs_bf = bf16(hs) ; h_bf = bf16(hs + pos)   (pos broadcast over batch)
__global__ __launch_bounds__(256)
void prep_inputs(const float* __restrict__ hs, const float* __restrict__ pos,
                 bf16* __restrict__ hsb, bf16* __restrict__ hb, int total) {
    int i = blockIdx.x * blockDim.x + threadIdx.x;
    if (i >= total) return;
    float v = hs[i];
    hsb[i] = (bf16)v;
    hb[i]  = (bf16)(v + pos[i % (NQ_ * EMBED_)]);
}

// Wt[c*256 + k] = bf16(W[k*C + c])  (K is always 256 here)
__global__ __launch_bounds__(256)
void prep_weight_t(const float* __restrict__ W, bf16* __restrict__ Wt, int Cc) {
    int i = blockIdx.x * blockDim.x + threadIdx.x;
    if (i >= 256 * Cc) return;
    int k = i / Cc, c = i % Cc;
    Wt[(size_t)c * 256 + k] = (bf16)W[i];
}

// ---------------- bf16 WMMA GEMM: C[M,NOUT] = A[M,256] @ Wt^T + bias ----------
// Block = 8 waves sharing one 64-column N-strip of W staged in LDS via the
// async-to-LDS path; each wave computes a 16x64 strip for a distinct M-tile.
// B fragments come from LDS (ds_load_b128, short fixed latency, bank-conflict
// free via row padding); A is preloaded into registers from global.
template<int NOUT>
__global__ __launch_bounds__(256)
void gemm_bf16_wmma(const bf16* __restrict__ A, const bf16* __restrict__ Wt,
                    const float* __restrict__ bias, float* __restrict__ C) {
    __shared__ bf16 Bt[64 * BPAD];               // 33 KB

    const int ntiles = NOUT / 64;
    const int tn = blockIdx.x % ntiles;
    const int tm = (blockIdx.x / ntiles) * 8 + (threadIdx.x >> 5);  // exact grid, no tail

    // ---- cooperative async stage of the 64x256 B strip into LDS ----
    {
        const bf16* bsrc = Wt + (size_t)tn * 64 * 256;
        const unsigned lds_base = (unsigned)(size_t)(void*)Bt;
#pragma unroll
        for (int r = 0; r < 8; ++r) {            // 2048 chunks / 256 threads
            int idx = r * 256 + threadIdx.x;
            int c = idx >> 5, i = idx & 31;      // col 0..63, 16B-chunk 0..31
            async_ld_b128(lds_base + (unsigned)(c * BPAD + i * 8) * 2,
                          bsrc + c * 256 + i * 8);
        }
        wait_asynccnt0();
        __syncthreads();
    }

    const int lane = threadIdx.x & 31;
    const int lm = lane & 15;
    const int hi = lane >> 4;

    // ---- preload full A slice: 16 x 16B chunks per lane (reused 4x each) ----
    // A fragment for K-step j: K = j*32 + {hi*16 .. +7} and {hi*16+8 .. +7}
    const bf16* arow = A + (size_t)(tm * 16 + lm) * 256;
    v8bf a0[8], a1[8];
#pragma unroll
    for (int j = 0; j < 8; ++j) {
        a0[j] = *(const v8bf*)(arow + j * 32 + hi * 16);
        a1[j] = *(const v8bf*)(arow + j * 32 + hi * 16 + 8);
    }

    // B fragment base pointers in LDS (lane lm = column, hi selects K half)
    const bf16* lb0 = &Bt[( 0 + lm) * BPAD + hi * 16];
    const bf16* lb1 = &Bt[(16 + lm) * BPAD + hi * 16];
    const bf16* lb2 = &Bt[(32 + lm) * BPAD + hi * 16];
    const bf16* lb3 = &Bt[(48 + lm) * BPAD + hi * 16];

    v8f c0 = {}, c1 = {}, c2 = {}, c3 = {};
#pragma unroll
    for (int j = 0; j < 8; ++j) {
        const int o = j * 32;
        v16bf a  = cat16(a0[j], a1[j]);
        v16bf f0 = cat16(*(const v8bf*)(lb0 + o), *(const v8bf*)(lb0 + o + 8));
        v16bf f1 = cat16(*(const v8bf*)(lb1 + o), *(const v8bf*)(lb1 + o + 8));
        v16bf f2 = cat16(*(const v8bf*)(lb2 + o), *(const v8bf*)(lb2 + o + 8));
        v16bf f3 = cat16(*(const v8bf*)(lb3 + o), *(const v8bf*)(lb3 + o + 8));
        c0 = __builtin_amdgcn_wmma_f32_16x16x32_bf16(false, a, false, f0, (short)0, c0, false, false);
        c1 = __builtin_amdgcn_wmma_f32_16x16x32_bf16(false, a, false, f1, (short)0, c1, false, false);
        c2 = __builtin_amdgcn_wmma_f32_16x16x32_bf16(false, a, false, f2, (short)0, c2, false, false);
        c3 = __builtin_amdgcn_wmma_f32_16x16x32_bf16(false, a, false, f3, (short)0, c3, false, false);
    }

    const int colbase = tn * 64 + lm;
    const float bb0 = bias[colbase +  0];
    const float bb1 = bias[colbase + 16];
    const float bb2 = bias[colbase + 32];
    const float bb3 = bias[colbase + 48];
#pragma unroll
    for (int r = 0; r < 8; ++r) {
        int row = tm * 16 + hi * 8 + r;   // D VGPR r -> M = r (lanes 0-15) / 8+r (lanes 16-31)
        float* crow = C + (size_t)row * NOUT + colbase;
        crow[0]  = c0[r] + bb0;
        crow[16] = c1[r] + bb1;
        crow[32] = c2[r] + bb2;
        crow[48] = c3[r] + bb3;
    }
}

// ---------------- sampling + softmax + weighted sum ----------------
// One wave per (n,q,head); lane = dh channel -> every gather is a coalesced 128B line.
__device__ __forceinline__ float gsample(const float* __restrict__ vb,
                                         int yi, int xi, int H, int W) {
    if (xi < 0 || xi >= W || yi < 0 || yi >= H) return 0.0f;
    return vb[(size_t)(yi * W + xi) * EMBED_];
}

__global__ __launch_bounds__(256)
void msda_sample(const float* __restrict__ values, const float* __restrict__ offb,
                 const float* __restrict__ attnb, const float* __restrict__ refp,
                 bf16* __restrict__ midb) {
    const int lane = threadIdx.x & 31;
    int wave = blockIdx.x * 8 + (threadIdx.x >> 5);
    if (wave >= N_ * NQ_ * HEADS_) return;
    const int h = wave % HEADS_;
    int t = wave / HEADS_;
    const int q = t % NQ_;
    const int n = t / NQ_;

    // softmax over LEVELS*POINTS = 16 logits (wave-uniform work)
    const float* al = attnb + (size_t)(n * NQ_ + q) * (HEADS_ * 16) + h * 16;
    float w16[16];
    float mx = -1e30f;
#pragma unroll
    for (int k = 0; k < 16; ++k) { w16[k] = al[k]; mx = fmaxf(mx, w16[k]); }
    float s = 0.0f;
#pragma unroll
    for (int k = 0; k < 16; ++k) { w16[k] = __expf(w16[k] - mx); s += w16[k]; }
    const float inv = 1.0f / s;

    // pts = (ref + off)[..., ::-1]  =>  x uses component 1, y uses component 0
    const float ry = refp[q * 2 + 0];
    const float rx = refp[q * 2 + 1];
    const float* offq = offb + (size_t)(n * NQ_ + q) * 256;

    const int Hs[4] = {64, 32, 16, 8};
    const int St[4] = {0, 4096, 5120, 5376};

    float acc = 0.0f;
#pragma unroll
    for (int l = 0; l < LEVELS_; ++l) {
        const int H = Hs[l], W = Hs[l], st = St[l];
        const float* vb = values + ((size_t)n * NQ_ + st) * EMBED_ + h * DH_ + lane;
#pragma unroll
        for (int p = 0; p < POINTS_; ++p) {
            const float* o = offq + ((l * HEADS_ + h) * POINTS_ + p) * 2;
            float y = ry + o[0];
            float x = rx + o[1];
            float fx = (x + 1.0f) * (W * 0.5f) - 0.5f;
            float fy = (y + 1.0f) * (H * 0.5f) - 0.5f;
            float x0f = floorf(fx), y0f = floorf(fy);
            float wx = fx - x0f, wy = fy - y0f;
            int x0 = (int)x0f, y0 = (int)y0f;
            float v00 = gsample(vb, y0,     x0,     H, W);
            float v01 = gsample(vb, y0,     x0 + 1, H, W);
            float v10 = gsample(vb, y0 + 1, x0,     H, W);
            float v11 = gsample(vb, y0 + 1, x0 + 1, H, W);
            float bil = v00 * (1.0f - wx) * (1.0f - wy) + v01 * wx * (1.0f - wy)
                      + v10 * (1.0f - wx) * wy          + v11 * wx * wy;
            acc += w16[l * POINTS_ + p] * inv * bil;
        }
    }
    midb[(size_t)(n * NQ_ + q) * EMBED_ + h * DH_ + lane] = (bf16)acc;
}

// ---------------- host-side launch ----------------
extern "C" void kernel_launch(void* const* d_in, const int* in_sizes, int n_in,
                              void* d_out, int out_size, void* d_ws, size_t ws_size,
                              hipStream_t stream) {
    const float* hs     = (const float*)d_in[0];
    const float* pos    = (const float*)d_in[1];
    const float* refp   = (const float*)d_in[2];
    // d_in[3] = spatial_shapes (compile-time constants here)
    const float* W_off  = (const float*)d_in[4];
    const float* b_off  = (const float*)d_in[5];
    const float* W_attn = (const float*)d_in[6];
    const float* b_attn = (const float*)d_in[7];
    const float* W_val  = (const float*)d_in[8];
    const float* b_val  = (const float*)d_in[9];
    const float* W_out  = (const float*)d_in[10];
    const float* b_out  = (const float*)d_in[11];

    char* ws = (char*)d_ws;
    size_t off = 0;
    auto alloc = [&](size_t bytes) {
        void* p = ws + off;
        off = (off + bytes + 255) & ~(size_t)255;
        return p;
    };
    bf16*  hs_bf   = (bf16*) alloc((size_t)M_ * 256 * 2);
    bf16*  h_bf    = (bf16*) alloc((size_t)M_ * 256 * 2);
    bf16*  mid_bf  = (bf16*) alloc((size_t)M_ * 256 * 2);
    bf16*  wt_val  = (bf16*) alloc(256 * 256 * 2);
    bf16*  wt_off  = (bf16*) alloc(256 * 256 * 2);
    bf16*  wt_attn = (bf16*) alloc(256 * 128 * 2);
    bf16*  wt_out  = (bf16*) alloc(256 * 256 * 2);
    float* valbuf  = (float*)alloc((size_t)M_ * 256 * 4);
    float* offbuf  = (float*)alloc((size_t)M_ * 256 * 4);
    float* attnbuf = (float*)alloc((size_t)M_ * 128 * 4);

    // 1) precision prep
    {
        int total = M_ * 256;
        prep_inputs<<<(total + 255) / 256, 256, 0, stream>>>(hs, pos, hs_bf, h_bf, total);
        prep_weight_t<<<(256 * 256 + 255) / 256, 256, 0, stream>>>(W_val,  wt_val,  256);
        prep_weight_t<<<(256 * 256 + 255) / 256, 256, 0, stream>>>(W_off,  wt_off,  256);
        prep_weight_t<<<(256 * 128 + 255) / 256, 256, 0, stream>>>(W_attn, wt_attn, 128);
        prep_weight_t<<<(256 * 256 + 255) / 256, 256, 0, stream>>>(W_out,  wt_out,  256);
    }

    const int mblocks = (M_ / 16) / 8;           // 170 blocks of 8 M-tiles
    // 2) values = hs @ W_val + b_val
    gemm_bf16_wmma<256><<<mblocks * 4, 256, 0, stream>>>(hs_bf, wt_val, b_val, valbuf);
    // 3) off = h @ W_off + b_off
    gemm_bf16_wmma<256><<<mblocks * 4, 256, 0, stream>>>(h_bf, wt_off, b_off, offbuf);
    // 4) attn logits = h @ W_attn + b_attn
    gemm_bf16_wmma<128><<<mblocks * 2, 256, 0, stream>>>(h_bf, wt_attn, b_attn, attnbuf);

    // 5) bilinear sampling + softmax + weighted sum -> mid (bf16)
    {
        int waves = N_ * NQ_ * HEADS_;           // 174080
        msda_sample<<<(waves + 7) / 8, 256, 0, stream>>>(valbuf, offbuf, attnbuf, refp, mid_bf);
    }

    // 6) out = mid @ W_out + b_out  (fp32 into d_out)
    gemm_bf16_wmma<256><<<mblocks * 4, 256, 0, stream>>>(mid_bf, wt_out, b_out, (float*)d_out);
}
